// SpectralConvND_55628416417814
// MI455X (gfx1250) — matrix operand order repaired and checked by reading, not compiled, not run on hardware
//
#include <hip/hip_runtime.h>
#include <hip/hip_bf16.h>

// ---------------------------------------------------------------------------
// SpectralConv (FNO) for MI455X / gfx1250, wave32.
// Truncated DFTs as GEMMs on v_wmma_f32_16x16x32_f16; x streamed HBM->LDS
// with the Tensor Data Mover (tensor_load_to_lds + s_wait_tensorcnt),
// double-buffered against WMMA compute.
// ---------------------------------------------------------------------------

typedef __attribute__((ext_vector_type(16))) _Float16 v16h;
typedef __attribute__((ext_vector_type(8)))  float    v8f;
typedef __attribute__((ext_vector_type(4)))  unsigned int u32x4;
typedef __attribute__((ext_vector_type(8)))  int      i32x8;
typedef __attribute__((ext_vector_type(4)))  int      i32x4;

#define DEVINL __device__ __forceinline__

union AFrag { v16h v; _Float16 h[16]; unsigned int u[8]; };
union CFrag { v8f  v; float f[8]; };

DEVINL v8f wmma16x16x32(const AFrag& a, const AFrag& b, v8f c) {
  return __builtin_amdgcn_wmma_f32_16x16x32_f16(
      false, a.v, false, b.v, (short)0, c, false, false);
}

// A-operand K column for slot v (half = lane>>4)
DEVINL int a_kcol(int v, int half) { return 2*(v&3) + ((v>>2)<<4) + (half<<3); }
// B-operand K row for slot v (khalf = lane>>4); always even
DEVINL int b_krow(int v, int khalf) { return 2*(v&3) + ((v>>2)<<4) + (khalf<<3); }

// ---- workspace layout (f16-element offsets) -------------------------------
// Pair-interleaved tables store element (k, col) at (k>>1)*2*NCOL + col*2 + (k&1)
// so a B-fragment slot (k,k+1 same col) is one aligned 32-bit load.
#define AFWD_OFF   0          // [64][512] row-major: rows 0..31 cos, 32..63 -sin
#define BC1B_OFF   32768      // pair-interleaved [512][32][2] from [1024][32] [cos; sin]
#define BS1B_OFF   65536      // pair-interleaved              [-sin; cos]
#define BC3_OFF    98304      // pair-interleaved [32][512][2] from [64][512] [w*cos; -w*sin]
#define BS3_OFF    131072     // pair-interleaved              [w*sin;  w*cos]
#define A4_OFF     163840     // [512][64] row-major [cos | -sin]
#define WF16_OFF   196608     // [1024 modes][32 o][64 = Wr(32)|Wi(32)]
#define WS_F16_ELEMS (196608 + 1024*32*64)
#define XM_OFF_BYTES  ((size_t)WS_F16_ELEMS * 2)                 // f32 [2][32c][8b][1024ij]
#define OM_OFF_BYTES  (XM_OFF_BYTES + (size_t)2*32*8*1024*4)     // f32 [2][32o][8b][1024ij]

// ---------------------------------------------------------------------------
// TDM: issue a 2D tensor_load_to_lds of a rows x 512 fp32 slab (row-major,
// stride 512) from global into LDS at byte offset lds_off.
// D# bitfields per CDNA5 ISA (group0 / group1).
// ---------------------------------------------------------------------------
DEVINL void tdm_load_slab(unsigned int lds_off, const float* gptr, unsigned int rows) {
  unsigned long long ga = (unsigned long long)(__UINTPTR_TYPE__)gptr;
  u32x4 g0;
  g0[0] = 1u;                                            // count=1 (valid), user mode
  g0[1] = lds_off;                                       // lds_addr (bytes)
  g0[2] = (unsigned int)(ga & 0xffffffffull);            // global_addr[31:0]
  g0[3] = (unsigned int)((ga >> 32) & 0x1ffffffull)      // global_addr[56:32]
        | (2u << 30);                                    // type = 2 ("image")
  i32x8 g1;
  g1[0] = (int)(2u << 16);       // workgroup_mask=0, data_size=2 (4 bytes)
  g1[1] = (int)(512u << 16);     // tensor_dim0[15:0] (bits 63:48) = 512
  g1[2] = (int)(512u << 16);     // tensor_dim0 hi=0 | tensor_dim1 lo16 (95:80) = 512
  g1[3] = (int)(512u << 16);     // tensor_dim1 hi=0 | tile_dim0 (127:112) = 512
  g1[4] = (int)rows;             // tile_dim1 (143:128)
  g1[5] = 512;                   // tensor_dim0_stride lo32 (191:160)
  g1[6] = 0;                     // stride hi | tensor_dim1_stride lo
  g1[7] = 0;
  i32x4 z4 = {};
  i32x8 z8 = {};
  // 6-arg toolchain variant: (g0, g1, g2, g3, g4, cpol)
  __builtin_amdgcn_tensor_load_to_lds(g0, g1, z4, z4, z8, 0);
}

// ===========================================================================
// K0: build twiddle tables (f16) in workspace
// ===========================================================================
__global__ void k0_tables(_Float16* ws16) {
  const float TWO_PI = 6.2831853071795864769f;
  int t0 = blockIdx.x * blockDim.x + threadIdx.x;
  int stride = gridDim.x * blockDim.x;

  // Afwd[r][h] row-major
  for (int idx = t0; idx < 64*512; idx += stride) {
    int r = idx >> 9, h = idx & 511, i = r & 31;
    float th = TWO_PI * (float)((i*h) & 511) * (1.0f/512.0f);
    ws16[AFWD_OFF + idx] = (_Float16)((r < 32) ? cosf(th) : -sinf(th));
  }
  // Forward col tables, pair-interleaved over k: addr = (k>>1)*64 + j*2 + (k&1)
  for (int idx = t0; idx < 1024*32; idx += stride) {
    int k = idx >> 5, j = idx & 31, w = k & 511;
    float th = TWO_PI * (float)((j*w) & 511) * (1.0f/512.0f);
    float c = cosf(th), s = sinf(th);
    int addr = (k>>1)*64 + j*2 + (k&1);
    ws16[BC1B_OFF + addr] = (_Float16)((k < 512) ?  c : s);
    ws16[BS1B_OFF + addr] = (_Float16)((k < 512) ? -s : c);
  }
  // Inverse col tables (Hermitian weight w_j), pair-interleaved:
  // addr = (k>>1)*1024 + w*2 + (k&1)
  for (int idx = t0; idx < 64*512; idx += stride) {
    int k = idx >> 9, w = idx & 511, j = k & 31;
    float wj = (j == 0) ? 1.0f : 2.0f;
    float th = TWO_PI * (float)((j*w) & 511) * (1.0f/512.0f);
    float c = wj * cosf(th), s = wj * sinf(th);
    int addr = (k>>1)*1024 + w*2 + (k&1);
    ws16[BC3_OFF + addr] = (_Float16)((k < 32) ? c : -s);
    ws16[BS3_OFF + addr] = (_Float16)((k < 32) ? s :  c);
  }
  // Inverse row matrix A4[h][k] row-major (A-operand: K-contiguous already)
  for (int idx = t0; idx < 512*64; idx += stride) {
    int h = idx >> 6, k = idx & 63, i = k & 31;
    float th = TWO_PI * (float)((i*h) & 511) * (1.0f/512.0f);
    ws16[A4_OFF + idx] = (_Float16)((k < 32) ? cosf(th) : -sinf(th));
  }
}

// ===========================================================================
// K1: Wc = proj @ coeff  ->  wf16[ij][o][Wr(32)|Wi(32)] f16 (one wave/mode)
// ===========================================================================
__global__ void __launch_bounds__(256) k1_weights(const float* __restrict__ proj,
                                                  const float* __restrict__ cr,
                                                  const float* __restrict__ ci,
                                                  _Float16* __restrict__ ws16) {
  int lane = threadIdx.x & 31;
  int wid  = threadIdx.x >> 5;
  int ij   = blockIdx.x * 8 + wid;
  int half = lane >> 4, m = lane & 15;
  int bn   = lane & 15, bkh = lane >> 4;
  _Float16* wout = ws16 + WF16_OFF + (size_t)ij * 2048;

  AFrag a[2];
#pragma unroll
  for (int mt = 0; mt < 2; ++mt)
#pragma unroll
    for (int v = 0; v < 8; ++v) {
      int k = a_kcol(v, half);
      int o = mt*16 + m;
      a[mt].h[2*v]   = (_Float16)proj[o*32 + k];
      a[mt].h[2*v+1] = (_Float16)proj[o*32 + k + 1];
    }

#pragma unroll
  for (int ct = 0; ct < 2; ++ct) {
#pragma unroll
    for (int ri = 0; ri < 2; ++ri) {
      const float* cf = ri ? ci : cr;
      AFrag b;
#pragma unroll
      for (int v = 0; v < 8; ++v) {
        int k = b_krow(v, bkh);
        int c = ct*16 + bn;
        b.h[2*v]   = (_Float16)cf[(size_t)(k  )*32768 + ij*32 + c];
        b.h[2*v+1] = (_Float16)cf[(size_t)(k+1)*32768 + ij*32 + c];
      }
#pragma unroll
      for (int mt = 0; mt < 2; ++mt) {
        CFrag d; d.v = {};
        d.v = wmma16x16x32(a[mt], b, d.v);
#pragma unroll
        for (int r = 0; r < 8; ++r) {
          int o = mt*16 + (lane>>4)*8 + r;
          int c = ct*16 + (lane&15);
          wout[o*64 + ri*32 + c] = (_Float16)d.f[r];
        }
      }
    }
  }
}

// ===========================================================================
// K2: forward truncated DFT, one workgroup (8 waves) per image (b,c).
//   x streamed HBM->LDS by TDM (32x512 fp32 slabs, double-buffered).
//   1a: Ycat(64x512) = Afwd(64x512) @ X(512x512)  -> LDS (f16)
//   1b: modes(2x32x32) = [Yr|Yi](32x1024) @ {Bc,Bs}(1024x32) -> xm (f32)
// ===========================================================================
__global__ void __launch_bounds__(256) k2_forward(const float* __restrict__ x,
                                                  const _Float16* __restrict__ ws16,
                                                  float* __restrict__ xm) {
  __shared__ alignas(16) _Float16 Y[64*512];     // 64 KB
  __shared__ alignas(16) float    Xs[2][32*512]; // 2 x 64 KB TDM staging
  int lane = threadIdx.x & 31;
  int wid  = threadIdx.x >> 5;
  int half = lane >> 4, m = lane & 15;
  int bn   = lane & 15, bkh = lane >> 4;
  const float* ximg = x + (size_t)blockIdx.x * (512*512);
  const _Float16* afwd = ws16 + AFWD_OFF;

  // ---- phase 1a: kc (K-slab) is the streaming loop -----------------------
  CFrag acc[4][4];                     // [mt][q] accumulators
#pragma unroll
  for (int mt = 0; mt < 4; ++mt)
#pragma unroll
    for (int q = 0; q < 4; ++q) acc[mt][q].v = {};

  if (wid == 0)
    tdm_load_slab((unsigned int)(__UINTPTR_TYPE__)&Xs[0][0], ximg, 32);

  for (int kc = 0; kc < 16; ++kc) {
    if (wid == 0) {
      if (kc + 1 < 16) {
        tdm_load_slab((unsigned int)(__UINTPTR_TYPE__)&Xs[(kc+1)&1][0],
                      ximg + (size_t)(kc+1)*32*512, 32);
        __builtin_amdgcn_s_wait_tensorcnt(1);  // slab kc done, kc+1 in flight
      } else {
        __builtin_amdgcn_s_wait_tensorcnt(0);  // final slab done
      }
    }
    __syncthreads();                   // slab kc ready for all waves
    const float* xs = Xs[kc & 1];

    for (int q = 0; q < 4; ++q) {
      int nt = wid + 8*q;
      AFrag bfr;
#pragma unroll
      for (int v = 0; v < 8; ++v) {    // B(k=h, n=w) from LDS slab, f32->f16
        int hrow = b_krow(v, bkh);     // row within slab
        int wcol = nt*16 + bn;
        bfr.h[2*v]   = (_Float16)xs[(hrow  )*512 + wcol];
        bfr.h[2*v+1] = (_Float16)xs[(hrow+1)*512 + wcol];
      }
#pragma unroll
      for (int mt = 0; mt < 4; ++mt) {
        AFrag afr;
#pragma unroll
        for (int v = 0; v < 8; ++v) {
          int k = a_kcol(v, half);
          afr.u[v] = *(const unsigned int*)(afwd + (mt*16+m)*512 + kc*32 + k);
        }
        acc[mt][q].v = wmma16x16x32(afr, bfr, acc[mt][q].v);
      }
    }
    __syncthreads();                   // everyone done with slab kc
  }

#pragma unroll
  for (int q = 0; q < 4; ++q) {
    int nt = wid + 8*q;
#pragma unroll
    for (int mt = 0; mt < 4; ++mt)
#pragma unroll
      for (int r = 0; r < 8; ++r) {
        int row = mt*16 + (lane>>4)*8 + r;
        int col = nt*16 + (lane&15);
        Y[row*512 + col] = (_Float16)acc[mt][q].f[r];
      }
  }
  __syncthreads();

  // ---- phase 1b ----------------------------------------------------------
  {
    int ri = wid & 1, nt = (wid>>1) & 1, mt = wid >> 2;  // 8 tiles, 1/wave
    const _Float16* btab = ws16 + (ri ? BS1B_OFF : BC1B_OFF);
    CFrag acc1; acc1.v = {};
    for (int kc = 0; kc < 32; ++kc) {  // K = 1024 = [Yr | Yi]
      AFrag afr, bfr;
#pragma unroll
      for (int v = 0; v < 8; ++v) {
        int kk = kc*32 + a_kcol(v, half);          // even; pair K-contiguous
        int irow = mt*16 + m;
        const _Float16* src = (kk < 512) ? &Y[irow*512 + kk]
                                         : &Y[(32+irow)*512 + (kk-512)];
        afr.u[v] = *(const unsigned int*)src;      // 32-bit LDS load
        int kb = kc*32 + b_krow(v, bkh);           // even
        int jc = nt*16 + bn;
        bfr.u[v] = *(const unsigned int*)(btab + (kb>>1)*64 + jc*2);
      }
      acc1.v = wmma16x16x32(afr, bfr, acc1.v);
    }
    int n = blockIdx.x, bidx = n >> 5, cidx = n & 31;   // n = b*32 + c
    size_t base = ((size_t)(ri*32 + cidx)*8 + bidx) * 1024;
#pragma unroll
    for (int r = 0; r < 8; ++r) {
      int irow = mt*16 + (lane>>4)*8 + r;
      int jcol = nt*16 + (lane&15);
      xm[base + irow*32 + jcol] = acc1.f[r];
    }
  }
}

// ===========================================================================
// K3: per-mode complex channel mix; one wave per mode (1024 modes).
// ===========================================================================
__global__ void __launch_bounds__(256) k3_mix(const _Float16* __restrict__ ws16,
                                              const float* __restrict__ xm,
                                              float* __restrict__ om) {
  int lane = threadIdx.x & 31;
  int wid  = threadIdx.x >> 5;
  int ij   = blockIdx.x * 8 + wid;
  int half = lane >> 4, m = lane & 15;
  int bn   = lane & 15, bkh = lane >> 4;
  const _Float16* wmat = ws16 + WF16_OFF + (size_t)ij * 2048;

  AFrag a[2][2];
#pragma unroll
  for (int mt = 0; mt < 2; ++mt)
#pragma unroll
    for (int kc = 0; kc < 2; ++kc)
#pragma unroll
      for (int v = 0; v < 8; ++v) {
        int k = kc*32 + a_kcol(v, half);
        a[mt][kc].u[v] = *(const unsigned int*)(wmat + (mt*16+m)*64 + k);
      }

  AFrag bfr[2][2];
#pragma unroll
  for (int which = 0; which < 2; ++which)
#pragma unroll
    for (int kc = 0; kc < 2; ++kc)
#pragma unroll
      for (int v = 0; v < 8; ++v)
#pragma unroll
        for (int e = 0; e < 2; ++e) {
          int k = kc*32 + b_krow(v, bkh) + e;
          float val = 0.0f;
          if (bn < 8) {
            int c = k & 31;
            int sel; float sgn = 1.0f;
            if (which == 0) { sel = (k < 32) ? 0 : 1; if (k >= 32) sgn = -1.0f; }
            else            { sel = (k < 32) ? 1 : 0; }
            val = sgn * xm[((size_t)(sel*32 + c)*8 + bn)*1024 + ij];
          }
          bfr[which][kc].h[2*v+e] = (_Float16)val;
        }

#pragma unroll
  for (int which = 0; which < 2; ++which)
#pragma unroll
    for (int mt = 0; mt < 2; ++mt) {
      CFrag acc; acc.v = {};
      acc.v = wmma16x16x32(a[mt][0], bfr[which][0], acc.v);
      acc.v = wmma16x16x32(a[mt][1], bfr[which][1], acc.v);
      int col = lane & 15;
      if (col < 8) {
#pragma unroll
        for (int r = 0; r < 8; ++r) {
          int o = mt*16 + (lane>>4)*8 + r;
          om[((size_t)(which*32 + o)*8 + col)*1024 + ij] = acc.f[r];
        }
      }
    }
}

// ===========================================================================
// K4: inverse transform, one workgroup per output image (b,o).
//   3a: Ucat(64x512) = [Mr|Mi](32x64)/(H*W) @ {Bc3,Bs3}(64x512) -> LDS
//   3b: out(512x512) = [Ch|-Sh](512x64) @ Ucat(64x512)          -> HBM
//   U stored pair-interleaved: (k,col) at (k>>1)*1024 + col*2 + (k&1)
// ===========================================================================
__global__ void __launch_bounds__(256) k4_inverse(const _Float16* __restrict__ ws16,
                                                  const float* __restrict__ om,
                                                  float* __restrict__ out) {
  __shared__ alignas(16) _Float16 U[64*512];   // 64 KB
  int lane = threadIdx.x & 31;
  int wid  = threadIdx.x >> 5;
  int half = lane >> 4, m = lane & 15;
  int bn   = lane & 15, bkh = lane >> 4;
  int img  = blockIdx.x;                       // (b, o)
  int bidx = img >> 5, o = img & 31;
  const float invHW = 1.0f / (512.0f * 512.0f);

  // ---- phase 3a: 128 tiles (ri x 32 nt x 2 mt), 16 per wave --------------
  for (int q = 0; q < 16; ++q) {
    int tt = wid*16 + q;
    int ri = tt & 1, nt = (tt>>1) & 31, mt = tt >> 6;
    const _Float16* btab = ws16 + (ri ? BS3_OFF : BC3_OFF);
    CFrag acc; acc.v = {};
#pragma unroll
    for (int kc = 0; kc < 2; ++kc) {           // K = 64 = [Mr | Mi]
      AFrag afr, bfr;
#pragma unroll
      for (int v = 0; v < 8; ++v) {
        int ka = kc*32 + a_kcol(v, half);
        int irow = mt*16 + m;
#pragma unroll
        for (int e = 0; e < 2; ++e) {
          int k = ka + e;
          int sel = k >> 5, j = k & 31;        // 0 -> Mr, 1 -> Mi
          afr.h[2*v+e] = (_Float16)(invHW *
              om[((size_t)(sel*32 + o)*8 + bidx)*1024 + irow*32 + j]);
        }
        int kb = kc*32 + b_krow(v, bkh);       // even
        bfr.u[v] = *(const unsigned int*)(btab + (kb>>1)*1024 + (nt*16+bn)*2);
      }
      acc.v = wmma16x16x32(afr, bfr, acc.v);
    }
#pragma unroll
    for (int r = 0; r < 8; ++r) {
      int krow = ri*32 + mt*16 + (lane>>4)*8 + r;
      int col  = nt*16 + (lane&15);
      U[(krow>>1)*1024 + col*2 + (krow&1)] = (_Float16)acc.f[r];
    }
  }
  __syncthreads();

  // ---- phase 3b: 1024 output tiles, 128 per wave -------------------------
  float* oimg = out + (size_t)img * (512*512);
  const _Float16* a4 = ws16 + A4_OFF;
  for (int q = 0; q < 128; ++q) {
    int tt = wid*128 + q;
    int nt = tt & 31, mt = tt >> 5;
    CFrag acc; acc.v = {};
#pragma unroll
    for (int kc = 0; kc < 2; ++kc) {           // K = 64
      AFrag afr, bfr;
#pragma unroll
      for (int v = 0; v < 8; ++v) {
        int ka = kc*32 + a_kcol(v, half);
        afr.u[v] = *(const unsigned int*)(a4 + (mt*16+m)*64 + ka);
        int kb = kc*32 + b_krow(v, bkh);       // even
        int wcol = nt*16 + bn;
        bfr.u[v] = *(const unsigned int*)(U + (kb>>1)*1024 + wcol*2);
      }
      acc.v = wmma16x16x32(afr, bfr, acc.v);
    }
#pragma unroll
    for (int r = 0; r < 8; ++r) {
      int hrow = mt*16 + (lane>>4)*8 + r;
      int wcol = nt*16 + (lane&15);
      oimg[(size_t)hrow*512 + wcol] = acc.f[r];
    }
  }
}

// ===========================================================================
extern "C" void kernel_launch(void* const* d_in, const int* in_sizes, int n_in,
                              void* d_out, int out_size, void* d_ws, size_t ws_size,
                              hipStream_t stream) {
  const float* x    = (const float*)d_in[0];   // (8, 32, 512, 512)
  const float* cr   = (const float*)d_in[1];   // (32, 32, 32, 32)
  const float* ci   = (const float*)d_in[2];   // (32, 32, 32, 32)
  const float* proj = (const float*)d_in[3];   // (32, 32)
  float* out = (float*)d_out;                  // (8, 32, 512, 512)

  _Float16* ws16 = (_Float16*)d_ws;
  float* xm = (float*)((char*)d_ws + XM_OFF_BYTES);
  float* om = (float*)((char*)d_ws + OM_OFF_BYTES);

  k0_tables <<<96, 256, 0, stream>>>(ws16);
  k1_weights<<<128, 256, 0, stream>>>(proj, cr, ci, ws16);
  k2_forward<<<256, 256, 0, stream>>>(x, ws16, xm);
  k3_mix    <<<128, 256, 0, stream>>>(ws16, xm, om);
  k4_inverse<<<256, 256, 0, stream>>>(ws16, om, out);
}
